// PureFIKANLinear_76854144794925
// MI455X (gfx1250) — matrix operand order repaired and checked by reading, not compiled, not run on hardware
//
#include <hip/hip_runtime.h>
#include <hip/hip_bf16.h>
#include <math.h>

// ---------------------------------------------------------------------------
// PureFIKANLinear for MI455X (gfx1250, wave32, WMMA)
//
// Fused formulation: out = [silu(x) | bases(x,d)]  @  [W_base | W_frac]^T
//   A: (4096 x 7168) bf16, pre-swizzled into WMMA A-fragment layout
//   B: (1024 x 7168) bf16, pre-swizzled into WMMA B-fragment layout
//   C: (4096 x 1024) f32 accumulated with v_wmma_f32_16x16x32_bf16
// Packed operands (70 MB) fit in the 192 MB L2 -> GEMM streams from L2.
// GEMM register tile: 4x4 WMMA tiles per wave -> 1 fragment load per 2 WMMAs.
// ---------------------------------------------------------------------------

typedef __bf16 v16bf __attribute__((ext_vector_type(16)));
typedef float  v8f   __attribute__((ext_vector_type(8)));

#define B_DIM 4096
#define F_DIM 1024
#define O_DIM 1024
#define G_NUM 5          // G
#define KTOT  7168       // F + F*(G+1)
#define KT32  224        // KTOT / 32 k-tiles
#define MT16  256        // B_DIM / 16 m-tiles
#define NT16  64         // O_DIM / 16 n-tiles

#define WSA_ELEMS ((size_t)B_DIM * KTOT)              // bf16
#define WSB_ELEMS ((size_t)O_DIM * KTOT)              // bf16
#define WSA_BYTES (WSA_ELEMS * 2)                     // 58,720,256
#define WSB_BYTES (WSB_ELEMS * 2)                     // 14,680,064

// ---------------------------------------------------------------------------
// Kernel 1: d = 0.99 * tanh(d_raw)     (F*G = 5120 elements)
// ---------------------------------------------------------------------------
__global__ void fikan_dtanh(const float* __restrict__ d_raw,
                            float* __restrict__ wsD) {
    int i = blockIdx.x * blockDim.x + threadIdx.x;
    if (i < F_DIM * G_NUM) wsD[i] = 0.99f * tanhf(d_raw[i]);
}

// ---------------------------------------------------------------------------
// WMMA fragment address helpers (ISA 7.12.2, wave32)
// A 16x32 bf16 tile: lane = (M&15) + 16*((K>>3)&1), half j = (K&7) + 8*(K>=16)
// B 32x16 bf16 tile: lane = (N&15) + 16*((K>>3)&1), half j = (K&7) + 8*(K>=16)
// Fragment storage: [tile][lane 0..31][halves 0..15] contiguous.
// ---------------------------------------------------------------------------
__device__ __forceinline__ size_t a_frag_idx(int b, int kglob) {
    int mtile = b >> 4, mrow = b & 15;
    int ktile = kglob >> 5, kk = kglob & 31;
    int lane  = mrow + (((kk >> 3) & 1) << 4);
    int j     = (kk & 7) + ((kk >> 4) << 3);
    return ((((size_t)ktile * MT16 + mtile) * 32 + lane) << 4) + j;
}
__device__ __forceinline__ size_t b_frag_idx(int n, int kglob) {
    int ntile = n >> 4, ncol = n & 15;
    int ktile = kglob >> 5, kk = kglob & 31;
    int lane  = ncol + (((kk >> 3) & 1) << 4);
    int j     = (kk & 7) + ((kk >> 4) << 3);
    return ((((size_t)ktile * NT16 + ntile) * 32 + lane) << 4) + j;
}

// ---------------------------------------------------------------------------
// Kernel 2: pack W = [base_weight | fractal_weight*fractal_scaler] -> bf16
// One thread per (n, k) element of the 1024 x 7168 operand.
// ---------------------------------------------------------------------------
__global__ void fikan_packW(const float* __restrict__ bw,
                            const float* __restrict__ fw,
                            const float* __restrict__ fs,
                            __bf16* __restrict__ wsB) {
    int gid = blockIdx.x * blockDim.x + threadIdx.x;
    int n = gid & (O_DIM - 1);
    int k = gid >> 10;
    if (k >= KTOT) return;
    float v;
    if (k < F_DIM) {
        v = bw[(size_t)n * F_DIM + k];
    } else {
        int kp = k - F_DIM;
        int f  = kp / 6;
        int g  = kp - f * 6;
        v = fw[((size_t)n * F_DIM + f) * 6 + g] * fs[(size_t)n * F_DIM + f];
    }
    wsB[b_frag_idx(n, k)] = (__bf16)v;
}

// ---------------------------------------------------------------------------
// Kernel 3: pack A = [silu(x) | fractal_bases(x,d)] -> bf16
// One thread per (b, f); computes silu + the DEPTH=8 recurrence (6-vector acc
// kept in registers with predicated updates -> v_cndmask, no scratch).
// ---------------------------------------------------------------------------
__global__ void fikan_packA(const float* __restrict__ x,
                            const float* __restrict__ wsD,
                            __bf16* __restrict__ wsA) {
    int gid = blockIdx.x * blockDim.x + threadIdx.x;
    int f = gid & (F_DIM - 1);
    int b = gid >> 10;
    if (b >= B_DIM) return;

    float xv = x[(size_t)b * F_DIM + f];

    // silu(x) -> k = f
    float sil = xv / (1.0f + __expf(-xv));
    wsA[a_frag_idx(b, f)] = (__bf16)sil;

    // d[f][0..4]
    float d0 = wsD[f * G_NUM + 0];
    float d1 = wsD[f * G_NUM + 1];
    float d2 = wsD[f * G_NUM + 2];
    float d3 = wsD[f * G_NUM + 3];
    float d4 = wsD[f * G_NUM + 4];

    // t = clip((x - A)/(B - A)) with A=-1, B=1
    float t = fminf(fmaxf((xv + 1.0f) * 0.5f, 0.0f), 1.0f);
    float P = 1.0f;
    float acc[6] = {0.f, 0.f, 0.f, 0.f, 0.f, 0.f};

    #pragma unroll
    for (int it = 0; it < 8; ++it) {
        float tf = t * 5.0f;
        int   ii = (int)floorf(tf);
        ii = ii < 0 ? 0 : (ii > 4 ? 4 : ii);
        float s  = tf - (float)ii;
        float di = (ii == 0) ? d0 : (ii == 1) ? d1 : (ii == 2) ? d2
                 : (ii == 3) ? d3 : d4;
        float pl = P * (1.0f - s);      // weight on node ii
        float pr = P * s;               // weight on node ii+1
        float bl = P * di * (1.0f - s); // bvec correction on node 0
        float br = P * di * s;          // bvec correction on node G
        #pragma unroll
        for (int m = 0; m < 6; ++m) {
            float add = 0.0f;
            if (m == 0) add -= bl;
            if (m == 5) add -= br;
            add += (m == ii)     ? pl : 0.0f;
            add += (m == ii + 1) ? pr : 0.0f;
            acc[m] += add;
        }
        t = s;
        P *= di;
    }
    // final hat(tK) term
    {
        float tf = t * 5.0f;
        int   ii = (int)floorf(tf);
        ii = ii < 0 ? 0 : (ii > 4 ? 4 : ii);
        float s  = tf - (float)ii;
        float pl = P * (1.0f - s);
        float pr = P * s;
        #pragma unroll
        for (int m = 0; m < 6; ++m) {
            float add = (m == ii) ? pl : 0.0f;
            add      += (m == ii + 1) ? pr : 0.0f;
            acc[m] += add;
        }
    }

    int kbase = F_DIM + f * 6;
    #pragma unroll
    for (int g = 0; g < 6; ++g)
        wsA[a_frag_idx(b, kbase + g)] = (__bf16)acc[g];
}

// ---------------------------------------------------------------------------
// Kernel 4: bf16 WMMA GEMM, C(4096x1024) = A(4096x7168) * B(7168x1024)
// 128 threads = 4 waves tiled 2(M) x 2(N); each wave computes a 4x4 grid of
// 16x16 WMMA tiles (64x64 output) -> 16 accumulators, 8 fragment loads per
// k-tile for 16 v_wmma ops (1 load : 2 wmma). Block covers 128x128 output.
// Grid (32, 8) = 256 workgroups.
// ---------------------------------------------------------------------------
__global__ void __launch_bounds__(128)
fikan_gemm(const __bf16* __restrict__ wsA,
           const __bf16* __restrict__ wsB,
           float* __restrict__ out) {
    const int lane = threadIdx.x & 31;
    const int wave = threadIdx.x >> 5;      // 0..3
    const int wm = wave & 1;                // 2 waves along M
    const int wn = wave >> 1;               // 2 waves along N
    const int mtile0 = blockIdx.x * 8 + wm * 4;   // 4 consecutive m-tiles
    const int ntile0 = blockIdx.y * 8 + wn * 4;   // 4 consecutive n-tiles

    v8f c[4][4];
    #pragma unroll
    for (int i = 0; i < 4; ++i)
        #pragma unroll
        for (int j = 0; j < 4; ++j)
            c[i][j] = (v8f){};

    for (int kt = 0; kt < KT32; ++kt) {
        const v16bf* Afr =
            (const v16bf*)(wsA + (((size_t)kt * MT16 + mtile0) << 9));
        const v16bf* Bfr =
            (const v16bf*)(wsB + (((size_t)kt * NT16 + ntile0) << 9));

        v16bf a[4], b[4];
        #pragma unroll
        for (int i = 0; i < 4; ++i) a[i] = Afr[i * 32 + lane];
        #pragma unroll
        for (int j = 0; j < 4; ++j) b[j] = Bfr[j * 32 + lane];

        if (kt + 1 < KT32) {
            // pull next k-tile toward the WGP -> global_prefetch_b8
            __builtin_prefetch(
                (const void*)(wsA + (((size_t)(kt + 1) * MT16 + mtile0) << 9) +
                              (size_t)lane * 16), 0, 3);
            __builtin_prefetch(
                (const void*)(wsB + (((size_t)(kt + 1) * NT16 + ntile0) << 9) +
                              (size_t)lane * 16), 0, 3);
        }

        #pragma unroll
        for (int i = 0; i < 4; ++i)
            #pragma unroll
            for (int j = 0; j < 4; ++j)
                c[i][j] = __builtin_amdgcn_wmma_f32_16x16x32_bf16(
                    false, a[i], false, b[j], (short)0, c[i][j], false, false);
    }

    // C/D layout: VGPR r, lane l -> row = r + 8*(l>=16), col = l&15
    const int hi8 = (lane >> 4) << 3;
    const int nc  = lane & 15;
    #pragma unroll
    for (int i = 0; i < 4; ++i) {
        const int mbase = (mtile0 + i) * 16 + hi8;
        #pragma unroll
        for (int j = 0; j < 4; ++j) {
            const int nbase = (ntile0 + j) * 16 + nc;
            #pragma unroll
            for (int r = 0; r < 8; ++r)
                out[(size_t)(mbase + r) * O_DIM + nbase] = c[i][j][r];
        }
    }
}

// ---------------------------------------------------------------------------
// Host launcher
// d_in: x(4096x1024 f32), base_weight(1024x1024 f32),
//       fractal_weight(1024x1024x6 f32), fractal_scaler(1024x1024 f32),
//       d_raw(1024x5 f32)
// d_out: 4096x1024 f32
// d_ws: >= 73,420,800 bytes (wsA bf16 | wsB bf16 | wsD f32)
// ---------------------------------------------------------------------------
extern "C" void kernel_launch(void* const* d_in, const int* in_sizes, int n_in,
                              void* d_out, int out_size, void* d_ws, size_t ws_size,
                              hipStream_t stream) {
    const float* x    = (const float*)d_in[0];
    const float* bw   = (const float*)d_in[1];
    const float* fw   = (const float*)d_in[2];
    const float* fs   = (const float*)d_in[3];
    const float* draw = (const float*)d_in[4];
    float* out = (float*)d_out;

    char* ws = (char*)d_ws;
    __bf16* wsA = (__bf16*)ws;
    __bf16* wsB = (__bf16*)(ws + WSA_BYTES);
    float*  wsD = (float*)(ws + WSA_BYTES + WSB_BYTES);

    // 1) d = 0.99 * tanh(d_raw)
    fikan_dtanh<<<(F_DIM * G_NUM + 255) / 256, 256, 0, stream>>>(draw, wsD);

    // 2) pack weights (O_DIM * KTOT threads)
    fikan_packW<<<(int)(((size_t)O_DIM * KTOT) / 256), 256, 0, stream>>>(
        bw, fw, fs, wsB);

    // 3) pack activations + fractal bases (B_DIM * F_DIM threads)
    fikan_packA<<<(int)(((size_t)B_DIM * F_DIM) / 256), 256, 0, stream>>>(
        x, wsD, wsA);

    // 4) fused WMMA GEMM (block 128 threads, 128x128 output per block)
    dim3 grid(B_DIM / 128, O_DIM / 128);
    fikan_gemm<<<grid, 128, 0, stream>>>(wsA, wsB, out);
}
